// MultiHeadAttentionWithContext_67783173865602
// MI455X (gfx1250) — compile-verified, matrix operand order checked by reading, not run on hardware
//
#include <hip/hip_runtime.h>
#include <hip/hip_bf16.h>

// ---------------------------------------------------------------------------
// MHA with context bias, CDNA5 (gfx1250) WMMA bf16 implementation.
// B=2, S=2048, D=1024, H=16, DK=64.
//   1) one-shot f32->bf16 convert of activations; transpose+convert of weights
//   2) WMMA GEMMs: A tile staged by TDM (tensor_load_to_lds, TENSORcnt),
//      B tile staged by global_load_async_to_lds_b128 (ASYNCcnt), both
//      double-buffered
//   3) flash attention (WMMA QK^T + online softmax + WMMA PV), bf16 out
//   4) WMMA output projection, then fused residual + LayerNorm
// ---------------------------------------------------------------------------

#define Bb 2
#define Ss 2048
#define Dd 1024
#define Hh 16
#define DKk 64

typedef __attribute__((ext_vector_type(8)))  __bf16 v8bf;
typedef __attribute__((ext_vector_type(16))) __bf16 v16bf;
typedef __attribute__((ext_vector_type(8)))  float  v8f;
typedef __attribute__((ext_vector_type(4)))  unsigned v4u;
typedef __attribute__((ext_vector_type(8)))  unsigned v8u;

__device__ inline unsigned short f2bf(float f) {
    unsigned u = __float_as_uint(f);
    u += 0x7FFFu + ((u >> 16) & 1u);          // round-to-nearest-even
    return (unsigned short)(u >> 16);
}

// Build a 16x32-K A/B fragment (bf16): low 8 halves at p, high 8 at p+16 elems.
__device__ inline v16bf mk16(const unsigned short* p) {
    v8bf lo = *(const v8bf*)p;
    v8bf hi = *(const v8bf*)(p + 16);
    return __builtin_shufflevector(lo, hi, 0,1,2,3,4,5,6,7,8,9,10,11,12,13,14,15);
}

__device__ inline v8f wmma_bf16(v16bf a, v16bf b, v8f c) {
    return __builtin_amdgcn_wmma_f32_16x16x32_bf16(false, a, false, b, (short)0, c,
                                                   false, false);
}

// 16B global -> LDS async copy (ASYNCcnt-tracked), per-lane.
__device__ inline void async_b128(unsigned lds_off, const unsigned short* gptr) {
    asm volatile("global_load_async_to_lds_b128 %0, %1, off"
                 :: "v"(lds_off), "v"(gptr) : "memory");
}
__device__ inline void wait_async0() {
    asm volatile("s_wait_asynccnt 0x0" ::: "memory");
}
__device__ inline unsigned lds_addr32(const void* p) {
    return (unsigned)(uintptr_t)p;            // low 32 bits of generic = LDS offset
}

// TDM 2-D tile load (TENSORcnt-tracked). Descriptor per ISA 08_async_tensor §8:
//   g1 packs data_size=2B, pad 16B after every 64B row (-> 80B LDS row stride),
//   tile = tileK x tileRows elements, tensor row stride = K elements.
__device__ inline void tdm_load_2d(unsigned lds_byte, unsigned long long gaddr,
                                   unsigned K, unsigned rowsTot,
                                   unsigned tileK, unsigned tileRows) {
    v4u g0;
    g0[0] = 1u;                                             // count=1 (user D#)
    g0[1] = lds_byte;                                       // lds_addr
    g0[2] = (unsigned)gaddr;                                // global_addr lo
    g0[3] = (unsigned)((gaddr >> 32) & 0x01FFFFFFull) | (2u << 30); // hi | type=2
    v8u g1;
    g1[0] = (1u << 16) | (1u << 20) | (3u << 22) | (3u << 25);
    //       data=2B     pad_en       intv=16DW    amount=4DW (16B)
    g1[1] = (K & 0xFFFFu) << 16;                            // tensor_dim0 lo16
    g1[2] = ((K >> 16) & 0xFFFFu) | ((rowsTot & 0xFFFFu) << 16); // dim0 hi|dim1 lo
    g1[3] = ((rowsTot >> 16) & 0xFFFFu) | (tileK << 16);    // dim1 hi | tile_dim0
    g1[4] = tileRows;                                       // tile_dim1 (dim2=0)
    g1[5] = K;                                              // dim0_stride lo32
    g1[6] = 0u;                                             // stride hi | d1stride
    g1[7] = 0u;
    asm volatile("tensor_load_to_lds %0, %1" :: "s"(g0), "s"(g1) : "memory");
}

// ---------------------------------------------------------------------------
// f32 -> bf16 bulk convert (vectorized 4-wide)
// ---------------------------------------------------------------------------
__global__ __launch_bounds__(256)
void cvt_kernel(const float* __restrict__ in, unsigned short* __restrict__ out) {
    const int i = blockIdx.x * 256 + threadIdx.x;
    const float4 v = ((const float4*)in)[i];
    unsigned u0 = (unsigned)f2bf(v.x) | ((unsigned)f2bf(v.y) << 16);
    unsigned u1 = (unsigned)f2bf(v.z) | ((unsigned)f2bf(v.w) << 16);
    ((uint2*)out)[i] = make_uint2(u0, u1);
}

// ---------------------------------------------------------------------------
// W[K,N] f32 -> Wt[N,K] bf16 (LDS-tiled transpose, both sides coalesced)
// ---------------------------------------------------------------------------
__global__ __launch_bounds__(256)
void transpose_cvt_kernel(const float* __restrict__ W, unsigned short* __restrict__ Wt,
                          int K, int N) {
    __shared__ float tile[32][33];
    const int bk = blockIdx.x * 32, bn = blockIdx.y * 32;
    const int tx = threadIdx.x & 31, ty = threadIdx.x >> 5;   // 32 x 8
    #pragma unroll
    for (int i = 0; i < 32; i += 8)
        tile[ty + i][tx] = W[(size_t)(bk + ty + i) * N + bn + tx];
    __syncthreads();
    #pragma unroll
    for (int i = 0; i < 32; i += 8)
        Wt[(size_t)(bn + ty + i) * K + bk + tx] = f2bf(tile[tx][ty + i]);
}

// ---------------------------------------------------------------------------
// ctx = market_context @ Wc + bc    (2 x 1024) -- tiny, scalar FMA is fine
// ---------------------------------------------------------------------------
__global__ __launch_bounds__(256)
void ctx_kernel(const float* __restrict__ mc, const float* __restrict__ Wc,
                const float* __restrict__ bc, float* __restrict__ ctxb) {
    int idx = blockIdx.x * 256 + threadIdx.x;       // 0..2047
    int b = idx >> 10, n = idx & 1023;
    float acc = bc[n];
    for (int k = 0; k < Dd; ++k)
        acc += mc[b * Dd + k] * Wc[(size_t)k * Dd + n];
    ctxb[idx] = acc;
}

// ---------------------------------------------------------------------------
// WMMA GEMM: out = Xb[M,K](bf16) @ Wt[N,K](bf16)^T + bias (+ optional ctx).
//   mode 0: f32 output [M,N]
//   mode 1: bf16 head-split output [B,H,S,DK]
//   mode 2: mode 1 + ctx[(row/S)*D + col]
// Block tile 128x64, 8 waves x (32x32), K-step 32, double-buffered LDS:
//   A tile via TDM tensor_load_to_lds (one instruction, wave 0)
//   B tile via per-lane global_load_async_to_lds_b128
// ---------------------------------------------------------------------------
#define BM 128
#define BN 64
#define BK 32
#define LDT 40   // padded LDS stride (elems): 80B rows, 16B-aligned chunks

__global__ __launch_bounds__(256)
void gemm_bias_kernel(const unsigned short* __restrict__ Xb,
                      const unsigned short* __restrict__ Wt,
                      const float* __restrict__ bias, const float* __restrict__ ctxadd,
                      float* __restrict__ outF, unsigned short* __restrict__ outH,
                      int M, int N, int K, int mode) {
    __shared__ unsigned short Asm[2][BM * LDT];
    __shared__ unsigned short Bsm[2][BN * LDT];

    const int t = threadIdx.x;
    const int w = t >> 5, l = t & 31, lh = l & 15, half = l >> 4;
    const int waveM = w & 3, waveN = w >> 2;
    const int blockM = blockIdx.y * BM, blockN = blockIdx.x * BN;

    // B-tile copy slot: chunk t covers (row = t>>2, 16B piece = t&3).
    const int rB = t >> 2;
    const int cOff = (t & 3) * 8;                        // element offset in row
    const unsigned short* gB = Wt + (size_t)(blockN + rB) * K + cOff;
    const unsigned lB    = lds_addr32(&Bsm[0][rB * LDT + cOff]);
    const unsigned lAb   = lds_addr32(&Asm[0][0]);
    const unsigned bufA  = BM * LDT * 2;                 // bytes per buffer
    const unsigned bufB  = BN * LDT * 2;
    const unsigned long long gA =
        (unsigned long long)(uintptr_t)(Xb + (size_t)blockM * K);

    v8f acc[2][2] = {};

    // prologue: stage k0 = 0 into buffer 0
    if (t < 32)
        tdm_load_2d(lAb, gA, (unsigned)K, (unsigned)M, BK, BM);
    async_b128(lB, gB);
    wait_async0();
    if (t < 32) __builtin_amdgcn_s_wait_tensorcnt(0);
    __syncthreads();

    const int steps = K / BK;
    for (int it = 0; it < steps; ++it) {
        const int cur = it & 1;
        if (it + 1 < steps) {                            // prefetch next tile
            const int ko = (it + 1) * BK;
            if (t < 32)
                tdm_load_2d(lAb + (cur ^ 1) * bufA, gA + (size_t)ko * 2,
                            (unsigned)K, (unsigned)M, BK, BM);
            async_b128(lB + (cur ^ 1) * bufB, gB + ko);
        }

        v16bf af[2], bf[2];
        #pragma unroll
        for (int mt = 0; mt < 2; ++mt) {
            const int row = waveM * 32 + mt * 16 + lh;
            af[mt] = mk16(&Asm[cur][row * LDT + half * 8]);
        }
        #pragma unroll
        for (int nt = 0; nt < 2; ++nt) {
            const int n = waveN * 32 + nt * 16 + lh;
            bf[nt] = mk16(&Bsm[cur][n * LDT + half * 8]);
        }
        #pragma unroll
        for (int mt = 0; mt < 2; ++mt)
            #pragma unroll
            for (int nt = 0; nt < 2; ++nt)
                acc[mt][nt] = wmma_bf16(af[mt], bf[nt], acc[mt][nt]);

        wait_async0();
        if (t < 32) __builtin_amdgcn_s_wait_tensorcnt(0);
        __syncthreads();
    }

    // --- epilogue: C layout lane l holds col lh, VGPR r holds row r+8*half ---
    #pragma unroll
    for (int mt = 0; mt < 2; ++mt) {
        #pragma unroll
        for (int nt = 0; nt < 2; ++nt) {
            const int col = blockN + waveN * 32 + nt * 16 + lh;
            const float bcol = bias[col];
            #pragma unroll
            for (int r = 0; r < 8; ++r) {
                const int row = blockM + waveM * 32 + mt * 16 + r + 8 * half;
                float v = acc[mt][nt][r] + bcol;
                if (mode == 2) v += ctxadd[(row >> 11) * Dd + col];
                if (mode == 0) {
                    outF[(size_t)row * N + col] = v;
                } else {
                    const int b = row >> 11, s = row & (Ss - 1);
                    const int h = col >> 6, dk = col & (DKk - 1);
                    outH[(((size_t)(b * Hh + h)) << 17) + ((size_t)s << 6) + dk] = f2bf(v);
                }
            }
        }
    }
}

// ---------------------------------------------------------------------------
// Flash attention: block = 8 waves x 16 q-rows = 128 q-rows of one (b,h).
// Per 32-key chunk: 4 WMMAs (Q@K^T), online softmax, P transposed through
// wave-local LDS, 4 WMMAs (P@V). Output bf16 (feeds Wo GEMM directly).
// ---------------------------------------------------------------------------
__global__ __launch_bounds__(256)
void attn_kernel(const unsigned short* __restrict__ Qp,
                 const unsigned short* __restrict__ Kp,
                 const unsigned short* __restrict__ Vp,
                 const int* __restrict__ mask,
                 unsigned short* __restrict__ ctx2) {
    __shared__ unsigned short Vt[DKk * 32];        // V tile transposed [dk][j]
    __shared__ unsigned short Pl[8][16 * 32];      // per-wave P scratch [16][32]

    const int t = threadIdx.x;
    const int w = t >> 5, l = t & 31, lh = l & 15, half = l >> 4;
    const int h = blockIdx.y, b = blockIdx.z;
    const int qbase = blockIdx.x * 128 + w * 16;
    const size_t headOff = ((size_t)(b * Hh + h)) << 17;   // * S * DK

    // Q A-fragments (16x32 each, DK split 0..31 / 32..63): row = qbase+lh
    const unsigned short* qrow = Qp + headOff + (size_t)(qbase + lh) * DKk;
    const v16bf Aq0 = mk16(qrow + half * 8);
    const v16bf Aq1 = mk16(qrow + 32 + half * 8);

    v8f O[4] = {};
    float m8[8], l8[8];
    #pragma unroll
    for (int r = 0; r < 8; ++r) { m8[r] = -3.0e38f; l8[r] = 0.0f; }

    for (int j = 0; j < Ss; j += 32) {
        __syncthreads();
        // --- stage V tile transposed: Vt[dk][jj] ---
        {
            const int dk = t & 63, jj0 = t >> 6;
            #pragma unroll
            for (int p = 0; p < 8; ++p) {
                const int jj = jj0 + p * 4;
                Vt[dk * 32 + jj] = Vp[headOff + (size_t)(j + jj) * DKk + dk];
            }
        }
        // --- scores: K rows map directly onto B-fragment layout ---
        const unsigned short* kr0 = Kp + headOff + (size_t)(j + lh) * DKk;
        const unsigned short* kr1 = kr0 + 16 * DKk;
        if (j + 32 < Ss)  // prefetch next K chunk (global_prefetch_b8)
            __builtin_prefetch(kr0 + 32 * DKk, 0, 1);
        const v16bf B00 = mk16(kr0 + half * 8);
        const v16bf B01 = mk16(kr0 + 32 + half * 8);
        const v16bf B10 = mk16(kr1 + half * 8);
        const v16bf B11 = mk16(kr1 + 32 + half * 8);
        const v8f z = {};
        v8f c0 = wmma_bf16(Aq1, B01, wmma_bf16(Aq0, B00, z));   // keys j..j+15
        v8f c1 = wmma_bf16(Aq1, B11, wmma_bf16(Aq0, B10, z));   // keys j+16..j+31

        // --- mask + scale + online softmax (row = r + 8*half, col = lane) ---
        #pragma unroll
        for (int r = 0; r < 8; ++r) {
            const int q = qbase + r + 8 * half;
            const size_t mrow = ((size_t)b * Ss + q) * Ss + j;
            float s0 = c0[r] * 0.125f;
            if (mask[mrow + lh] == 0) s0 = -1.0e9f;
            float s1 = c1[r] * 0.125f;
            if (mask[mrow + 16 + lh] == 0) s1 = -1.0e9f;

            float tmax = fmaxf(s0, s1);
            #pragma unroll
            for (int off = 1; off < 16; off <<= 1)
                tmax = fmaxf(tmax, __shfl_xor(tmax, off, 16));
            const float mnew  = fmaxf(m8[r], tmax);
            const float alpha = __expf(m8[r] - mnew);
            const float p0 = __expf(s0 - mnew);
            const float p1 = __expf(s1 - mnew);
            float rs = p0 + p1;
            #pragma unroll
            for (int off = 1; off < 16; off <<= 1)
                rs += __shfl_xor(rs, off, 16);
            l8[r] = l8[r] * alpha + rs;
            m8[r] = mnew;
            #pragma unroll
            for (int tt = 0; tt < 4; ++tt) O[tt][r] *= alpha;

            unsigned short* pw = &Pl[w][(r + 8 * half) * 32];
            pw[lh]      = f2bf(p0);
            pw[16 + lh] = f2bf(p1);
        }
        __syncthreads();   // Vt staged + own P visible

        // --- P (16x32) @ V-tile (32x16 per dk-slice): 4 WMMAs ---
        const v16bf Pa = mk16(&Pl[w][lh * 32 + half * 8]);
        #pragma unroll
        for (int tt = 0; tt < 4; ++tt) {
            const v16bf Vb = mk16(&Vt[(tt * 16 + lh) * 32 + half * 8]);
            O[tt] = wmma_bf16(Pa, Vb, O[tt]);
        }
    }

    // --- epilogue: normalize by l, write bf16 ctx2 [B,S,D] head-merged ---
    #pragma unroll
    for (int tt = 0; tt < 4; ++tt)
        #pragma unroll
        for (int r = 0; r < 8; ++r) {
            const int q = qbase + r + 8 * half;
            const int col = h * DKk + tt * 16 + lh;
            ctx2[((size_t)(b * Ss + q)) * Dd + col] = f2bf(O[tt][r] / l8[r]);
        }
}

// ---------------------------------------------------------------------------
// Residual + LayerNorm: one block per row of 1024.
// ---------------------------------------------------------------------------
__global__ __launch_bounds__(256)
void ln_kernel(const float* __restrict__ o, const float* __restrict__ q,
               const float* __restrict__ g, const float* __restrict__ be,
               float* __restrict__ out) {
    __shared__ float red[8];
    const int row = blockIdx.x, t = threadIdx.x, w = t >> 5, l = t & 31;
    const float* xr = o + (size_t)row * Dd;
    const float* qr = q + (size_t)row * Dd;
    float x[4];
    float s = 0.0f;
    #pragma unroll
    for (int i = 0; i < 4; ++i) { x[i] = xr[t + i * 256] + qr[t + i * 256]; s += x[i]; }
    #pragma unroll
    for (int off = 16; off; off >>= 1) s += __shfl_xor(s, off, 32);
    if (l == 0) red[w] = s;
    __syncthreads();
    float tot = red[0] + red[1] + red[2] + red[3] + red[4] + red[5] + red[6] + red[7];
    const float mu = tot * (1.0f / 1024.0f);
    float d[4], vs = 0.0f;
    #pragma unroll
    for (int i = 0; i < 4; ++i) { d[i] = x[i] - mu; vs += d[i] * d[i]; }
    #pragma unroll
    for (int off = 16; off; off >>= 1) vs += __shfl_xor(vs, off, 32);
    __syncthreads();
    if (l == 0) red[w] = vs;
    __syncthreads();
    float var = (red[0] + red[1] + red[2] + red[3] + red[4] + red[5] + red[6] + red[7])
                * (1.0f / 1024.0f);
    const float rstd = rsqrtf(var + 1e-5f);
    #pragma unroll
    for (int i = 0; i < 4; ++i) {
        const int c = t + i * 256;
        out[(size_t)row * Dd + c] = d[i] * rstd * g[c] + be[c];
    }
}

// ---------------------------------------------------------------------------
extern "C" void kernel_launch(void* const* d_in, const int* in_sizes, int n_in,
                              void* d_out, int out_size, void* d_ws, size_t ws_size,
                              hipStream_t stream) {
    const float* query = (const float*)d_in[0];
    const float* key_  = (const float*)d_in[1];
    const float* value = (const float*)d_in[2];
    const int*   mask  = (const int*)d_in[3];
    const float* mc    = (const float*)d_in[4];
    const float* Wq = (const float*)d_in[5],  *bq = (const float*)d_in[6];
    const float* Wk = (const float*)d_in[7],  *bk = (const float*)d_in[8];
    const float* Wv = (const float*)d_in[9],  *bv = (const float*)d_in[10];
    const float* Wo = (const float*)d_in[11], *bo = (const float*)d_in[12];
    const float* Wc = (const float*)d_in[13], *bc = (const float*)d_in[14];
    const float* ln_g = (const float*)d_in[15], *ln_b = (const float*)d_in[16];
    float* out = (float*)d_out;

    const int    M = Bb * Ss;                              // 4096
    const size_t actElems  = (size_t)M * Dd;               // 4M
    const size_t headElems = (size_t)Bb * Hh * Ss * DKk;   // 4M
    const size_t wElems    = (size_t)Dd * Dd;              // 1M

    char* wsp = (char*)d_ws;
    unsigned short* Qb  = (unsigned short*)wsp; wsp += actElems * 2;   // bf16 inputs
    unsigned short* Kb  = (unsigned short*)wsp; wsp += actElems * 2;
    unsigned short* Vb  = (unsigned short*)wsp; wsp += actElems * 2;
    unsigned short* Wqt = (unsigned short*)wsp; wsp += wElems * 2;     // bf16 W^T
    unsigned short* Wkt = (unsigned short*)wsp; wsp += wElems * 2;
    unsigned short* Wvt = (unsigned short*)wsp; wsp += wElems * 2;
    unsigned short* Wot = (unsigned short*)wsp; wsp += wElems * 2;
    unsigned short* Qp  = (unsigned short*)wsp; wsp += headElems * 2;  // projected
    unsigned short* Kp  = (unsigned short*)wsp; wsp += headElems * 2;
    unsigned short* Vp  = (unsigned short*)wsp; wsp += headElems * 2;
    float* ctxb = (float*)wsp; wsp += (size_t)Bb * Dd * 4;
    unsigned short* ctx2 = (unsigned short*)wsp; wsp += actElems * 2;
    float* out2 = (float*)wsp; wsp += actElems * 4;

    const dim3 gg(Dd / BN, M / BM);                        // (16, 32)
    const dim3 tg(Dd / 32, Dd / 32);                       // (32, 32)

    // one-shot precision/layout prep
    cvt_kernel<<<actElems / 4 / 256, 256, 0, stream>>>(query, Qb);
    cvt_kernel<<<actElems / 4 / 256, 256, 0, stream>>>(key_,  Kb);
    cvt_kernel<<<actElems / 4 / 256, 256, 0, stream>>>(value, Vb);
    transpose_cvt_kernel<<<tg, 256, 0, stream>>>(Wq, Wqt, Dd, Dd);
    transpose_cvt_kernel<<<tg, 256, 0, stream>>>(Wk, Wkt, Dd, Dd);
    transpose_cvt_kernel<<<tg, 256, 0, stream>>>(Wv, Wvt, Dd, Dd);
    transpose_cvt_kernel<<<tg, 256, 0, stream>>>(Wo, Wot, Dd, Dd);
    ctx_kernel<<<(Bb * Dd) / 256, 256, 0, stream>>>(mc, Wc, bc, ctxb);

    // projections (WMMA, TDM + async-LDS staged)
    gemm_bias_kernel<<<gg, 256, 0, stream>>>(Qb, Wqt, bq, nullptr, nullptr, Qp,
                                             M, Dd, Dd, 1);
    gemm_bias_kernel<<<gg, 256, 0, stream>>>(Kb, Wkt, bk, ctxb, nullptr, Kp,
                                             M, Dd, Dd, 2);
    gemm_bias_kernel<<<gg, 256, 0, stream>>>(Vb, Wvt, bv, nullptr, nullptr, Vp,
                                             M, Dd, Dd, 1);
    // attention
    attn_kernel<<<dim3(Ss / 128, Hh, Bb), 256, 0, stream>>>(Qp, Kp, Vp, mask, ctx2);
    // output projection + residual/LN
    gemm_bias_kernel<<<gg, 256, 0, stream>>>(ctx2, Wot, bo, nullptr, out2, nullptr,
                                             M, Dd, Dd, 0);
    ln_kernel<<<M, 256, 0, stream>>>(out2, query, ln_g, ln_b, out);
}
